// GraphTransformerRegressor_76630806495962
// MI455X (gfx1250) — matrix operand order repaired and checked by reading, not compiled, not run on hardware
//
#include <hip/hip_runtime.h>
#include <math.h>

#define HID 128
#define NH  4
#define CH  32

typedef __attribute__((ext_vector_type(16))) _Float16 v16h;
typedef __attribute__((ext_vector_type(8)))  float    v8f;

__device__ __forceinline__ float gelu_exact(float x) {
  return 0.5f * x * (1.0f + erff(x * 0.70710678118654752f));
}

// ---------------- fp32 -> fp16 elementwise ----------------
__global__ void cvt_f32_f16(const float* __restrict__ src,
                            _Float16* __restrict__ dst, int n) {
  int t = blockIdx.x * blockDim.x + threadIdx.x;
  if (t < n) dst[t] = (_Float16)src[t];
}

// ---- pack 4 weight mats [K][128] fp32 -> fp16 transposed [4][128][K] ----
__global__ void convert_w4_t(const float* __restrict__ wq, const float* __restrict__ wk,
                             const float* __restrict__ wv, const float* __restrict__ ws,
                             _Float16* __restrict__ Wh, int K) {
  int t = blockIdx.x * blockDim.x + threadIdx.x;
  int tot = 4 * HID * K;
  if (t >= tot) return;
  int p = t / (HID * K);
  int r = t - p * (HID * K);
  int n = r / K;
  int k = r - n * K;
  const float* W = (p == 0) ? wq : (p == 1) ? wk : (p == 2) ? wv : ws;
  Wh[((size_t)p * HID + n) * K + k] = (_Float16)W[(size_t)k * HID + n];
}

// ---------------- WMMA projection GEMM ----------------
// Y[p][r][n] = sum_k Xh[r][k] * W_p[k][n] + bias_p[n]
// grid.x = ceil(N/16) row tiles, grid.y = 4 projections (q,k,v,s)
// block = 256 = 8 wave32s; wave w computes columns [16w, 16w+16)
__global__ __launch_bounds__(256) void proj_gemm_wmma(
    const _Float16* __restrict__ Xh,  // [N][K]
    const _Float16* __restrict__ Wh,  // [4][128][K] (transposed fp16)
    const float* __restrict__ b0, const float* __restrict__ b1,
    const float* __restrict__ b2, const float* __restrict__ b3,
    float* __restrict__ Y,            // [4][N][128]
    int Nrows, int K) {
  const int lane = threadIdx.x & 31;
  const int wv_  = threadIdx.x >> 5;
  const int proj = blockIdx.y;
  const int row0 = blockIdx.x * 16;
  const int m  = lane & 15;   // A row / B,C column within tile
  const int kh = lane >> 4;   // half-wave selector
  int rr = row0 + m; if (rr > Nrows - 1) rr = Nrows - 1;
  const _Float16* Xrow = Xh + (size_t)rr * K;
  const _Float16* Wcol = Wh + ((size_t)proj * HID + (wv_ * 16 + m)) * K;
  const float* bias = (proj == 0) ? b0 : (proj == 1) ? b1 : (proj == 2) ? b2 : b3;
  v8f acc = {0.f, 0.f, 0.f, 0.f, 0.f, 0.f, 0.f, 0.f};
  for (int k0 = 0; k0 < K; k0 += 32) {
    v16h a, b;
#pragma unroll
    for (int j = 0; j < 8; ++j) {
      // A (16x32 f16): lanes0-15 K={0..7,16..23}, lanes16-31 K={8..15,24..31}
      const int Ka = ((j < 4) ? (2 * j) : (16 + 2 * (j - 4))) + 8 * kh;
      a[2 * j]     = Xrow[k0 + Ka];
      a[2 * j + 1] = Xrow[k0 + Ka + 1];
      // B (32x16 f16): lane = column; lanes0-15 K={0..7,16..23}, lanes16-31 K={8..15,24..31}
      const int Kb = ((j < 4) ? (2 * j) : (16 + 2 * (j - 4))) + 8 * kh;
      b[2 * j]     = Wcol[k0 + Kb];
      b[2 * j + 1] = Wcol[k0 + Kb + 1];
    }
    acc = __builtin_amdgcn_wmma_f32_16x16x32_f16(
        /*neg_a=*/false, a, /*neg_b=*/false, b,
        /*c_mod=*/(short)0, acc, /*reuse_a=*/false, /*reuse_b=*/false);
  }
  const int n = wv_ * 16 + m;
  const float bn = bias[n];
#pragma unroll
  for (int i = 0; i < 8; ++i) {   // D: VGPR i -> row i + 8*kh
    int r = row0 + i + 8 * kh;
    if (r < Nrows) Y[((size_t)proj * Nrows + r) * HID + n] = acc[i] + bn;
  }
}

// ---------------- CSR build (incoming edges per destination) ----------------
__global__ void deg_count(const int* __restrict__ ei, int* __restrict__ deg, int Ecnt) {
  int e = blockIdx.x * blockDim.x + threadIdx.x;
  if (e < Ecnt) atomicAdd(&deg[ei[Ecnt + e]], 1);
}

// single-block exclusive scan over deg -> rowptr (+ cursor copy); N up to a few M is fine
__global__ __launch_bounds__(256) void scan_rowptr(const int* __restrict__ deg,
                                                   int* __restrict__ rowptr,
                                                   int* __restrict__ cursor, int Nn) {
  __shared__ int wsum[8];
  __shared__ int carry;
  const int lane = threadIdx.x & 31;
  const int wid  = threadIdx.x >> 5;
  if (threadIdx.x == 0) carry = 0;
  __syncthreads();
  for (int base = 0; base < Nn; base += 256) {
    int i = base + (int)threadIdx.x;
    int d = (i < Nn) ? deg[i] : 0;
    // inclusive scan within wave32
    int v = d;
#pragma unroll
    for (int off = 1; off < 32; off <<= 1) {
      int nv = __shfl_up(v, off, 32);
      if (lane >= off) v += nv;
    }
    if (lane == 31) wsum[wid] = v;
    __syncthreads();
    if (wid == 0) {
      int s = (lane < 8) ? wsum[lane] : 0;
#pragma unroll
      for (int off = 1; off < 8; off <<= 1) {
        int ns = __shfl_up(s, off, 32);
        if (lane >= off) s += ns;
      }
      if (lane < 8) wsum[lane] = s;
    }
    __syncthreads();
    int excl = carry + (wid ? wsum[wid - 1] : 0) + (v - d);
    if (i < Nn) { rowptr[i] = excl; cursor[i] = excl; }
    __syncthreads();
    if (threadIdx.x == 0) carry += wsum[7];
    __syncthreads();
  }
  if (threadIdx.x == 0) rowptr[Nn] = carry;
}

__global__ void fill_elist(const int* __restrict__ ei, int* __restrict__ cursor,
                           int* __restrict__ elist, int Ecnt) {
  int e = blockIdx.x * blockDim.x + threadIdx.x;
  if (e >= Ecnt) return;
  int src = ei[e], dst = ei[Ecnt + e];
  int pos = atomicAdd(&cursor[dst], 1);
  elist[pos] = src;
}

// ------- fused attention (online softmax) + beta gate + LN + GELU -------
// One wave32 per destination node. Lane l owns channels 4l..4l+3, so the
// 8-lane sub-group (l>>3) == head h covers channels h*32..h*32+31 and the
// per-head q.k dot is a 3-step shfl_xor reduction. Running max/sum/acc in
// registers (flash-attention style) -> zero atomics, zero scratch passes.
__global__ __launch_bounds__(256) void attn_combine(
    const float* __restrict__ Q, const float* __restrict__ Kf,
    const float* __restrict__ V, const float* __restrict__ Xr,
    const int* __restrict__ rowptr, const int* __restrict__ elist,
    const float* __restrict__ Wbeta, const float* __restrict__ lng,
    const float* __restrict__ lnb, float* __restrict__ h32,
    _Float16* __restrict__ hh, int Nrows) {
  const int lane = threadIdx.x & 31;
  const int node = blockIdx.x * 8 + (threadIdx.x >> 5);
  if (node >= Nrows) return;
  const int c0 = lane * 4;  // first of 4 contiguous channels this lane owns
  const float4 q4 = *(const float4*)(Q + (size_t)node * HID + c0);
  const int beg = rowptr[node], end = rowptr[node + 1];
  float mrun = -3.0e38f, lsum = 0.f;
  float4 acc = {0.f, 0.f, 0.f, 0.f};
  for (int j = beg; j < end; ++j) {
    int src = elist[j];
    if (j + 1 < end) {  // hide gather latency: global_prefetch_b8 next rows
      int nsrc = elist[j + 1];
      __builtin_prefetch(Kf + (size_t)nsrc * HID + c0, 0, 0);
      __builtin_prefetch(V  + (size_t)nsrc * HID + c0, 0, 0);
    }
    float4 k4 = *(const float4*)(Kf + (size_t)src * HID + c0);
    float4 v4 = *(const float4*)(V  + (size_t)src * HID + c0);
    float part = q4.x * k4.x + q4.y * k4.y + q4.z * k4.z + q4.w * k4.w;
    part += __shfl_xor(part, 1, 32);
    part += __shfl_xor(part, 2, 32);
    part += __shfl_xor(part, 4, 32);          // per-head dot, all 8 lanes agree
    float alpha = part * 0.17677669529663687f; // 1/sqrt(32)
    float mnew  = fmaxf(mrun, alpha);
    float scale = expf(mrun - mnew);           // first iter: exp(-huge) == 0
    float p     = expf(alpha - mnew);
    lsum = lsum * scale + p;
    acc.x = acc.x * scale + p * v4.x;
    acc.y = acc.y * scale + p * v4.y;
    acc.z = acc.z * scale + p * v4.z;
    acc.w = acc.w * scale + p * v4.w;
    mrun = mnew;
  }
  const float inv = (lsum > 0.f) ? 1.f / lsum : 0.f;
  float4 o4 = {acc.x * inv, acc.y * inv, acc.z * inv, acc.w * inv};
  const float4 xr4 = *(const float4*)(Xr + (size_t)node * HID + c0);
  // beta gate: sigmoid([out, x_r, out - x_r] . Wbeta)
  const float4 wb0 = *(const float4*)(Wbeta + c0);
  const float4 wb1 = *(const float4*)(Wbeta + HID + c0);
  const float4 wb2 = *(const float4*)(Wbeta + 2 * HID + c0);
  float z = o4.x * wb0.x + o4.y * wb0.y + o4.z * wb0.z + o4.w * wb0.w
          + xr4.x * wb1.x + xr4.y * wb1.y + xr4.z * wb1.z + xr4.w * wb1.w
          + (o4.x - xr4.x) * wb2.x + (o4.y - xr4.y) * wb2.y
          + (o4.z - xr4.z) * wb2.z + (o4.w - xr4.w) * wb2.w;
#pragma unroll
  for (int off = 16; off > 0; off >>= 1) z += __shfl_xor(z, off, 32);
  const float beta = 1.f / (1.f + expf(-z));
  float hv[4];
  hv[0] = beta * xr4.x + (1.f - beta) * o4.x;
  hv[1] = beta * xr4.y + (1.f - beta) * o4.y;
  hv[2] = beta * xr4.z + (1.f - beta) * o4.z;
  hv[3] = beta * xr4.w + (1.f - beta) * o4.w;
  float s = hv[0] + hv[1] + hv[2] + hv[3];
  float sq = hv[0] * hv[0] + hv[1] * hv[1] + hv[2] * hv[2] + hv[3] * hv[3];
#pragma unroll
  for (int off = 16; off > 0; off >>= 1) {
    s  += __shfl_xor(s, off, 32);
    sq += __shfl_xor(sq, off, 32);
  }
  const float mu  = s * (1.f / HID);
  const float var = sq * (1.f / HID) - mu * mu;
  const float rst = rsqrtf(var + 1e-5f);
  const float4 g4 = *(const float4*)(lng + c0);
  const float4 b4 = *(const float4*)(lnb + c0);
  float y0 = gelu_exact((hv[0] - mu) * rst * g4.x + b4.x);
  float y1 = gelu_exact((hv[1] - mu) * rst * g4.y + b4.y);
  float y2 = gelu_exact((hv[2] - mu) * rst * g4.z + b4.z);
  float y3 = gelu_exact((hv[3] - mu) * rst * g4.w + b4.w);
  float4 yo = {y0, y1, y2, y3};
  *(float4*)(h32 + (size_t)node * HID + c0) = yo;
  _Float16* hp = hh + (size_t)node * HID + c0;
  hp[0] = (_Float16)y0; hp[1] = (_Float16)y1;
  hp[2] = (_Float16)y2; hp[3] = (_Float16)y3;
}

// ---------------- mean pool + MLP head ----------------
__global__ void pool_add(const float* __restrict__ h32, const int* __restrict__ batch,
                         float* __restrict__ gpool, int Nrows) {
  int t = blockIdx.x * blockDim.x + threadIdx.x;
  if (t >= Nrows * HID) return;
  int n = t >> 7, j = t & (HID - 1);
  atomicAdd(&gpool[(size_t)batch[n] * HID + j], h32[t]);
}
__global__ void pool_cnt(const int* __restrict__ batch, float* __restrict__ cnt, int Nrows) {
  int n = blockIdx.x * blockDim.x + threadIdx.x;
  if (n < Nrows) atomicAdd(&cnt[batch[n]], 1.f);
}
__global__ void final_mlp(const float* __restrict__ gpool, const float* __restrict__ cnt,
                          const float* __restrict__ W1, const float* __restrict__ b1,
                          const float* __restrict__ W2, const float* __restrict__ b2,
                          float* __restrict__ out) {
  __shared__ float gm[HID];
  __shared__ float red[HID];
  int g = blockIdx.x, j = threadIdx.x;
  float c = cnt[g]; if (c < 1.f) c = 1.f;
  gm[j] = gpool[(size_t)g * HID + j] / c;
  __syncthreads();
  float acc = b1[j];
  for (int k = 0; k < HID; ++k) acc += gm[k] * W1[(size_t)k * HID + j];
  red[j] = gelu_exact(acc) * W2[j];
  __syncthreads();
  for (int s = HID / 2; s > 0; s >>= 1) {
    if (j < s) red[j] += red[j + s];
    __syncthreads();
  }
  if (j == 0) out[g] = red[0] + b2[0];
}

extern "C" void kernel_launch(void* const* d_in, const int* in_sizes, int n_in,
                              void* d_out, int out_size, void* d_ws, size_t ws_size,
                              hipStream_t stream) {
  (void)n_in; (void)ws_size;
  const float* x     = (const float*)d_in[0];
  const int*   ei    = (const int*)d_in[1];
  const int*   batch = (const int*)d_in[2];
  const float* l0_Wq = (const float*)d_in[3];
  const float* l0_bq = (const float*)d_in[4];
  const float* l0_Wk = (const float*)d_in[5];
  const float* l0_bk = (const float*)d_in[6];
  const float* l0_Wv = (const float*)d_in[7];
  const float* l0_bv = (const float*)d_in[8];
  const float* l0_Ws = (const float*)d_in[9];
  const float* l0_bs = (const float*)d_in[10];
  const float* l0_Wb = (const float*)d_in[11];
  const float* l0_g  = (const float*)d_in[12];
  const float* l0_b  = (const float*)d_in[13];
  const float* Wq    = (const float*)d_in[14];
  const float* bq    = (const float*)d_in[15];
  const float* Wk    = (const float*)d_in[16];
  const float* bk    = (const float*)d_in[17];
  const float* Wv    = (const float*)d_in[18];
  const float* bv    = (const float*)d_in[19];
  const float* Ws_   = (const float*)d_in[20];
  const float* bs    = (const float*)d_in[21];
  const float* Wb    = (const float*)d_in[22];
  const float* lg    = (const float*)d_in[23];
  const float* lb    = (const float*)d_in[24];
  const float* W1    = (const float*)d_in[25];
  const float* b1    = (const float*)d_in[26];
  const float* W2    = (const float*)d_in[27];
  const float* b2    = (const float*)d_in[28];
  float* out = (float*)d_out;

  const int Nn   = in_sizes[2];       // N nodes
  const int INf  = in_sizes[0] / Nn;  // input feature dim (64)
  const int Ecnt = in_sizes[1] / 2;   // E edges
  const int G    = out_size;          // graphs

  char* base = (char*)d_ws;
  size_t off = 0;
  auto alloc = [&](size_t bytes) -> char* {
    char* p = base + off;
    off += (bytes + 255) & ~(size_t)255;
    return p;
  };
  _Float16* hh     = (_Float16*)alloc((size_t)Nn * HID * sizeof(_Float16));
  _Float16* Wh     = (_Float16*)alloc((size_t)4 * HID * HID * sizeof(_Float16));
  float*    Y4     = (float*)alloc((size_t)4 * Nn * HID * sizeof(float));
  float*    h32    = (float*)alloc((size_t)Nn * HID * sizeof(float));
  int*      deg    = (int*)alloc((size_t)Nn * sizeof(int));
  int*      rowptr = (int*)alloc((size_t)(Nn + 1) * sizeof(int));
  int*      cursor = (int*)alloc((size_t)Nn * sizeof(int));
  int*      elist  = (int*)alloc((size_t)Ecnt * sizeof(int));
  float*    gp     = (float*)alloc((size_t)G * HID * sizeof(float));
  float*    cnt    = (float*)alloc((size_t)G * sizeof(float));

  const int TB = 256;

  // ---- CSR of incoming edges, built once per call ----
  hipMemsetAsync(deg, 0, (size_t)Nn * sizeof(int), stream);
  deg_count<<<(Ecnt + TB - 1) / TB, TB, 0, stream>>>(ei, deg, Ecnt);
  scan_rowptr<<<1, TB, 0, stream>>>(deg, rowptr, cursor, Nn);
  fill_elist<<<(Ecnt + TB - 1) / TB, TB, 0, stream>>>(ei, cursor, elist, Ecnt);

  // layer-0 input features to fp16
  {
    int tot = Nn * INf;
    cvt_f32_f16<<<(tot + TB - 1) / TB, TB, 0, stream>>>(x, hh, tot);
  }

  for (int layer = 0; layer < 3; ++layer) {
    const float *pWq, *pWk, *pWv, *pWs, *pbq, *pbk, *pbv, *pbs, *pWb, *pg, *pb;
    int K;
    if (layer == 0) {
      K = INf;
      pWq = l0_Wq; pWk = l0_Wk; pWv = l0_Wv; pWs = l0_Ws;
      pbq = l0_bq; pbk = l0_bk; pbv = l0_bv; pbs = l0_bs;
      pWb = l0_Wb; pg = l0_g; pb = l0_b;
    } else {
      int i = layer - 1; K = HID;
      pWq = Wq  + (size_t)i * HID * HID; pWk = Wk  + (size_t)i * HID * HID;
      pWv = Wv  + (size_t)i * HID * HID; pWs = Ws_ + (size_t)i * HID * HID;
      pbq = bq + (size_t)i * HID; pbk = bk + (size_t)i * HID;
      pbv = bv + (size_t)i * HID; pbs = bs + (size_t)i * HID;
      pWb = Wb + (size_t)i * 3 * HID; pg = lg + (size_t)i * HID; pb = lb + (size_t)i * HID;
    }
    { int tot = 4 * HID * K;
      convert_w4_t<<<(tot + TB - 1) / TB, TB, 0, stream>>>(pWq, pWk, pWv, pWs, Wh, K); }
    dim3 gg((Nn + 15) / 16, 4);
    proj_gemm_wmma<<<gg, 256, 0, stream>>>(hh, Wh, pbq, pbk, pbv, pbs, Y4, Nn, K);
    const float* Qp = Y4;
    const float* Kp = Y4 + (size_t)Nn * HID;
    const float* Vp = Y4 + (size_t)2 * Nn * HID;
    const float* Sp = Y4 + (size_t)3 * Nn * HID;
    attn_combine<<<(Nn + 7) / 8, 256, 0, stream>>>(Qp, Kp, Vp, Sp, rowptr, elist,
                                                   pWb, pg, pb, h32, hh, Nn);
  }

  hipMemsetAsync(gp,  0, (size_t)G * HID * sizeof(float), stream);
  hipMemsetAsync(cnt, 0, (size_t)G * sizeof(float), stream);
  { int tot = Nn * HID;
    pool_add<<<(tot + TB - 1) / TB, TB, 0, stream>>>(h32, batch, gp, Nn); }
  pool_cnt<<<(Nn + TB - 1) / TB, TB, 0, stream>>>(batch, cnt, Nn);
  final_mlp<<<G, HID, 0, stream>>>(gp, cnt, W1, b1, W2, b2, out);
}
